// NestedGraphTitanNL_2534030704729
// MI455X (gfx1250) — matrix operand order repaired and checked by reading, not compiled, not run on hardware
//
#include <hip/hip_runtime.h>
#include <hip/hip_bf16.h>
#include <math.h>

// ---------------------------------------------------------------------------
// MI455X (gfx1250) implementation.
// GEMMs: TDM (tensor_load_to_lds) stages A/W tiles into LDS; bf16 fragments
//        read via ds_load; f32 accumulate via v_wmma_f32_16x16x32_bf16.
// Scan:  per-sequence LDS-resident 96x96 state, VALU + prefetch.
// ---------------------------------------------------------------------------

typedef __attribute__((ext_vector_type(16))) __bf16 v16bf;
typedef __attribute__((ext_vector_type(8)))  float  v8f;
typedef __attribute__((ext_vector_type(4)))  unsigned int v4u;
typedef __attribute__((ext_vector_type(8)))  int v8i;
typedef __attribute__((ext_vector_type(4)))  int v4i;

#ifndef __has_builtin
#define __has_builtin(x) 0
#endif
#if __has_builtin(__builtin_amdgcn_tensor_load_to_lds)
#define HAVE_TDM 1
#else
#define HAVE_TDM 0
#endif

static constexpr int ROWS = 32768;   // B*S*N = 32*256*4
static constexpr int DMODEL = 96;

__device__ __forceinline__ float sigmoidf_(float x) { return 1.f / (1.f + __expf(-x)); }
__device__ __forceinline__ float siluf_(float x)    { return x * sigmoidf_(x); }

// ---------------------------------------------------------------------------
// TDM: async 2-D bf16 tile load Global -> LDS (D# per CDNA5 ISA ch.8).
//   group0: count=1 | lds_addr | global_addr | type=2
//   group1: data_size=2B | tensor_dim0/1 | tile_dim0/1 | dim0_stride
// Issued once (caller restricts to one wave); EXEC is ignored by TDM.
// ---------------------------------------------------------------------------
__device__ __forceinline__ void tdm_load_2d_bf16(unsigned lds_off, const void* gaddr,
                                                 unsigned dim0, unsigned dim1,
                                                 unsigned tile0, unsigned tile1,
                                                 unsigned stride0)
{
#if HAVE_TDM
    unsigned long long ga = (unsigned long long)(size_t)gaddr;
    v4u g0;
    g0[0] = 1u;                                              // count=1 (valid user D#)
    g0[1] = lds_off;                                         // lds_addr [63:32]
    g0[2] = (unsigned)(ga & 0xffffffffu);                    // global_addr[31:0]
    g0[3] = (unsigned)((ga >> 32) & 0x1ffffffu)              // global_addr[56:32]
          | 0x80000000u;                                     // type=2 ("image") [127:126]
    v8i g1;
    g1[0] = (int)(1u << 16);                                 // data_size=1 -> 2 bytes
    g1[1] = (int)((dim0 & 0xffffu) << 16);                   // tensor_dim0[15:0] @ [63:48]
    g1[2] = (int)((dim0 >> 16) | ((dim1 & 0xffffu) << 16));  // dim0[31:16] | dim1[15:0]
    g1[3] = (int)((dim1 >> 16) | (tile0 << 16));             // dim1[31:16] | tile_dim0
    g1[4] = (int)(tile1 & 0xffffu);                          // tile_dim1 (tile_dim2=0)
    g1[5] = (int)stride0;                                    // dim0_stride[31:0]
    g1[6] = 0;                                               // stride0[47:32] | stride1 lo
    g1[7] = 0;
    v4i z4 = {0, 0, 0, 0};
#if defined(__clang_major__) && (__clang_major__ >= 23)
    v8i z8 = {0, 0, 0, 0, 0, 0, 0, 0};
    __builtin_amdgcn_tensor_load_to_lds(g0, g1, z4, z4, z8, 0);
#else
    __builtin_amdgcn_tensor_load_to_lds(g0, g1, z4, z4, 0);
#endif
#endif
}

// ---------------------------------------------------------------------------
// f32 -> bf16 convert (weights + activations staging)
// ---------------------------------------------------------------------------
__global__ void f32_to_bf16_kernel(const float* __restrict__ src, __bf16* __restrict__ dst, int n) {
    int i = blockIdx.x * blockDim.x + threadIdx.x;
    if (i < n) dst[i] = (__bf16)src[i];
}

// ---------------------------------------------------------------------------
// WMMA GEMM:  C[M,N] = A[M,K] @ W[N,K]^T (+bias) (+silu)
// block = 128 threads (4 waves). Block tile: 32 rows x 64 cols.
// TDM stages A tile (32xK) and W slice (64xK) into LDS; wave w owns n-tile
// n0 = blockIdx.y*64 + w*16 and runs two 16x16x32 WMMA chains (m-subtiles).
// Fragment layouts per CDNA5 ISA 7.12.2 (16-bit A 16x32, B 32x16).
// K must be a multiple of 32 and <= 288.
// ---------------------------------------------------------------------------
__global__ void gemm_wmma_bf16(const __bf16* __restrict__ A, const __bf16* __restrict__ W,
                               const float* __restrict__ bias,
                               float* __restrict__ Cf, __bf16* __restrict__ Cb,
                               int M, int N, int K, int act)
{
    __shared__ __bf16 lds[32 * 288 + 64 * 288];          // [A tile | W tile] = 55296 B
    __bf16* lA = lds;
    __bf16* lW = lds + 32 * 288;
    const unsigned LW_OFF = 32u * 288u * 2u;             // byte offset of lW in LDS

    const int lane = threadIdx.x & 31;
    const int wid  = threadIdx.x >> 5;
    const int m0 = blockIdx.x * 32;
    const int n0 = blockIdx.y * 64 + wid * 16;

#if HAVE_TDM
    if (threadIdx.x < 32) {                              // one wave issues both DMAs
        tdm_load_2d_bf16(0u,     A + (size_t)m0 * K,              K, (unsigned)M, K, 32u, K);
        tdm_load_2d_bf16(LW_OFF, W + (size_t)(blockIdx.y * 64) * K, K, (unsigned)N, K, 64u, K);
#if __has_builtin(__builtin_amdgcn_s_wait_tensorcnt)
        __builtin_amdgcn_s_wait_tensorcnt(0);
#endif
    }
#else
    for (int idx = threadIdx.x; idx < 32 * K; idx += (int)blockDim.x)
        lA[idx] = A[(size_t)(m0 + idx / K) * K + (idx % K)];
    for (int idx = threadIdx.x; idx < 64 * K; idx += (int)blockDim.x) {
        const int row = blockIdx.y * 64 + idx / K;
        lW[idx] = (row < N) ? W[(size_t)row * K + (idx % K)] : (__bf16)0.f;
    }
#endif
    __syncthreads();
    // The TDM DMA wrote LDS behind the compiler's back; escape the tile
    // pointers through an opaque writer so the ds_loads below are not folded
    // to undef ("never-stored" shared memory).
    asm volatile("" : : "r"(lA), "r"(lW) : "memory");

    const int ks8  = (lane >> 4) << 3;                   // A frag k sub-offset 0 / 8
    const int ko16 = (lane >> 4) << 4;                   // unused (B frag also uses ks8)
    const int ar   = lane & 15;                          // local A row (subtile 0)
    const int wr   = wid * 16 + (lane & 15);             // local W row (= output col)

    v8f acc0 = {}, acc1 = {};
    for (int kb = 0; kb < K; kb += 32) {
        v16bf b, a0, a1;
        const __bf16* wp = lW + wr * K + kb;
#pragma unroll
        for (int j = 0; j < 8; ++j) { b[j] = wp[ks8 + j]; b[8 + j] = wp[16 + ks8 + j]; }
        const __bf16* a0p = lA + ar * K + kb;
        const __bf16* a1p = a0p + 16 * K;
#pragma unroll
        for (int j = 0; j < 8; ++j) { a0[j] = a0p[ks8 + j]; a0[8 + j] = a0p[16 + ks8 + j]; }
#pragma unroll
        for (int j = 0; j < 8; ++j) { a1[j] = a1p[ks8 + j]; a1[8 + j] = a1p[16 + ks8 + j]; }
        acc0 = __builtin_amdgcn_wmma_f32_16x16x32_bf16(false, a0, false, b, (short)0, acc0, false, false);
        acc1 = __builtin_amdgcn_wmma_f32_16x16x32_bf16(false, a1, false, b, (short)0, acc1, false, false);
    }

    if (n0 < N) {
        const int ncol = n0 + (lane & 15);
        const float bv = bias ? bias[ncol] : 0.f;
        const int mb = m0 + ((lane >> 4) << 3);          // VGPR r -> M = mb + r
#pragma unroll
        for (int r = 0; r < 8; ++r) {
            float v0 = acc0[r] + bv;
            float v1 = acc1[r] + bv;
            if (act == 1) { v0 = siluf_(v0); v1 = siluf_(v1); }
            const size_t i0 = (size_t)(mb + r) * N + ncol;
            const size_t i1 = (size_t)(mb + 16 + r) * N + ncol;
            if (Cf) { Cf[i0] = v0; Cf[i1] = v1; }
            if (Cb) { Cb[i0] = (__bf16)v0; Cb[i1] = (__bf16)v1; }
        }
    }
}

// ---------------------------------------------------------------------------
// Embed: x[r,34] -> silu(x@W1^T+b1)[64] -> (@W2^T+b2)[96] + pos -> LN -> hF,hB
// one block (96 threads) per row
// ---------------------------------------------------------------------------
__global__ void embed_kernel(const float* __restrict__ x,
                             const float* __restrict__ W1, const float* __restrict__ b1,
                             const float* __restrict__ W2, const float* __restrict__ b2,
                             const float* __restrict__ lnw, const float* __restrict__ lnb,
                             const float* __restrict__ pe,
                             float* __restrict__ hF, __bf16* __restrict__ hB)
{
    __shared__ float mid[64];
    __shared__ float val[96];
    __shared__ float stats[2];
    const int r = blockIdx.x;            // ((b*256+s)*4+n)
    const int tid = threadIdx.x;         // 0..95
    const int s = (r >> 2) & 255;
    const float* xr = x + (size_t)r * 34;

    if (tid < 64) {
        float acc = b1[tid];
        const float* w = W1 + tid * 34;
#pragma unroll
        for (int j = 0; j < 34; ++j) acc += w[j] * xr[j];
        mid[tid] = siluf_(acc);
    }
    __syncthreads();
    {
        float acc = b2[tid];
        const float* w = W2 + tid * 64;
#pragma unroll
        for (int j = 0; j < 64; ++j) acc += w[j] * mid[j];
        val[tid] = acc + pe[s * 96 + tid];
    }
    __syncthreads();
    if (tid == 0) {
        float m = 0.f;
        for (int j = 0; j < 96; ++j) m += val[j];
        m *= (1.f / 96.f);
        float v = 0.f;
        for (int j = 0; j < 96; ++j) { float d = val[j] - m; v += d * d; }
        stats[0] = m; stats[1] = rsqrtf(v * (1.f / 96.f) + 1e-5f);
    }
    __syncthreads();
    float y = (val[tid] - stats[0]) * stats[1] * lnw[tid] + lnb[tid];
    hF[(size_t)r * 96 + tid] = y;
    hB[(size_t)r * 96 + tid] = (__bf16)y;
}

// ---------------------------------------------------------------------------
// wave-per-row  out = LN(a (+res)) * w + b   (96 cols, 3 per lane, wave32)
// ---------------------------------------------------------------------------
__global__ void add_ln_kernel(const float* __restrict__ a, const float* res,
                              const float* __restrict__ lnw, const float* __restrict__ lnb,
                              float* outF, __bf16* outB, int rows)
{
    const int wave = (blockIdx.x * blockDim.x + threadIdx.x) >> 5;
    const int lane = threadIdx.x & 31;
    if (wave >= rows) return;
    const float* ar = a + (size_t)wave * 96;
    float v0 = ar[lane], v1 = ar[lane + 32], v2 = ar[lane + 64];
    if (res) {
        const float* rr = res + (size_t)wave * 96;
        v0 += rr[lane]; v1 += rr[lane + 32]; v2 += rr[lane + 64];
    }
    float s = v0 + v1 + v2;
#pragma unroll
    for (int o = 16; o > 0; o >>= 1) s += __shfl_xor(s, o, 32);
    const float mean = s * (1.f / 96.f);
    float d0 = v0 - mean, d1 = v1 - mean, d2 = v2 - mean;
    float vv = d0 * d0 + d1 * d1 + d2 * d2;
#pragma unroll
    for (int o = 16; o > 0; o >>= 1) vv += __shfl_xor(vv, o, 32);
    const float rstd = rsqrtf(vv * (1.f / 96.f) + 1e-5f);
    float y0 = d0 * rstd * lnw[lane]      + lnb[lane];
    float y1 = d1 * rstd * lnw[lane + 32] + lnb[lane + 32];
    float y2 = d2 * rstd * lnw[lane + 64] + lnb[lane + 64];
    if (outF) {
        float* po = outF + (size_t)wave * 96;
        po[lane] = y0; po[lane + 32] = y1; po[lane + 64] = y2;
    }
    if (outB) {
        __bf16* pb = outB + (size_t)wave * 96;
        pb[lane] = (__bf16)y0; pb[lane + 32] = (__bf16)y1; pb[lane + 64] = (__bf16)y2;
    }
}

// wave-per-row L2-normalize (k -> k/max(||k||,1e-12)), in place
__global__ void knorm_kernel(float* __restrict__ k, int rows)
{
    const int wave = (blockIdx.x * blockDim.x + threadIdx.x) >> 5;
    const int lane = threadIdx.x & 31;
    if (wave >= rows) return;
    float* kr = k + (size_t)wave * 96;
    float v0 = kr[lane], v1 = kr[lane + 32], v2 = kr[lane + 64];
    float ss = v0 * v0 + v1 * v1 + v2 * v2;
#pragma unroll
    for (int o = 16; o > 0; o >>= 1) ss += __shfl_xor(ss, o, 32);
    const float inv = 1.f / fmaxf(sqrtf(ss), 1e-12f);
    kr[lane] = v0 * inv; kr[lane + 32] = v1 * inv; kr[lane + 64] = v2 * inv;
}

// ---------------------------------------------------------------------------
// eta / alpha heads: 96 -> 24 (silu) -> 1 ; thread per row
// ---------------------------------------------------------------------------
__global__ void eta_al_kernel(const float* __restrict__ h,
                              const float* W1e, const float* b1e, const float* w2e, const float* b2e,
                              const float* W1a, const float* b1a, const float* w2a, const float* b2a,
                              float* __restrict__ eta, float* __restrict__ al, int rows)
{
    const int r = blockIdx.x * blockDim.x + threadIdx.x;
    if (r >= rows) return;
    const float* x = h + (size_t)r * 96;
    float ae = b2e[0], aa = b2a[0];
    for (int o = 0; o < 24; ++o) {
        float se = b1e[o], sa = b1a[o];
        const float* we = W1e + o * 96;
        const float* wa = W1a + o * 96;
        for (int j = 0; j < 96; ++j) { float xv = x[j]; se += we[j] * xv; sa += wa[j] * xv; }
        ae += w2e[o] * siluf_(se);
        aa += w2a[o] * siluf_(sa);
    }
    eta[r] = 0.1f * sigmoidf_(ae) + 0.01f;
    al[r]  = 0.5f * sigmoidf_(aa) + 0.5f;
}

// ---------------------------------------------------------------------------
// Delta scan: 1 block per sequence g (128 seqs x 256 steps), 96 threads,
// state M[96x96] in LDS (pitch 97 vs 64 banks).
//   out_t = M q_t ; M <- al*M + eta*(vh - M k)*k^T
// Final M written straight to d_out slice.
// ---------------------------------------------------------------------------
__global__ void delta_scan_kernel(const float* __restrict__ q, const float* __restrict__ kn,
                                  const float* __restrict__ vh, const float* __restrict__ eta,
                                  const float* __restrict__ al,
                                  float* __restrict__ outF, __bf16* __restrict__ outB,
                                  float* __restrict__ Mout)
{
    __shared__ float M[96 * 97];
    __shared__ float qs[96], ks[96], vs[96];
    __shared__ float sc[2];
    const int g = blockIdx.x;
    const int i = threadIdx.x;              // row of M
    for (int idx = i; idx < 96 * 97; idx += 96) M[idx] = 0.f;
    __syncthreads();
    float* Mi = &M[i * 97];
    for (int t = 0; t < 256; ++t) {
        const size_t r = (size_t)g * 256 + t;
        qs[i] = q[r * 96 + i];
        ks[i] = kn[r * 96 + i];
        vs[i] = vh[r * 96 + i];
        if (i == 0) { sc[0] = eta[r]; sc[1] = al[r]; }
        if (t + 1 < 256 && i == 0) {        // prefetch next step's vectors
            __builtin_prefetch(q  + (r + 1) * 96, 0, 1);
            __builtin_prefetch(kn + (r + 1) * 96, 0, 1);
            __builtin_prefetch(vh + (r + 1) * 96, 0, 1);
        }
        __syncthreads();
        float dq = 0.f, dk = 0.f;
#pragma unroll 4
        for (int j = 0; j < 96; ++j) { float m = Mi[j]; dq += m * qs[j]; dk += m * ks[j]; }
        outF[r * 96 + i] = dq;
        outB[r * 96 + i] = (__bf16)dq;
        const float coef = sc[0] * (vs[i] - dk);
        const float a = sc[1];
#pragma unroll 4
        for (int j = 0; j < 96; ++j) Mi[j] = a * Mi[j] + coef * ks[j];
        __syncthreads();
    }
    for (int idx = i; idx < 96 * 96; idx += 96)
        Mout[(size_t)g * 9216 + idx] = M[(idx / 96) * 97 + (idx % 96)];
}

// ---------------------------------------------------------------------------
// CMS level aggregation: softmax(level_weights) . outs
// ---------------------------------------------------------------------------
__global__ void cms_combine_kernel(const float* __restrict__ o0, const float* __restrict__ o1,
                                   const float* __restrict__ o2, const float* __restrict__ lw,
                                   float* __restrict__ agg, int n)
{
    const int i = blockIdx.x * blockDim.x + threadIdx.x;
    if (i >= n) return;
    const float w0 = lw[0], w1 = lw[1], w2 = lw[2];
    const float mx = fmaxf(w0, fmaxf(w1, w2));
    const float e0 = __expf(w0 - mx), e1 = __expf(w1 - mx), e2 = __expf(w2 - mx);
    const float inv = 1.f / (e0 + e1 + e2);
    agg[i] = (e0 * o0[i] + e1 * o1[i] + e2 * o2[i]) * inv;
}

// ---------------------------------------------------------------------------
// Regime + head: one block (128 threads) per batch b. Entire N=4 mixing,
// gating, attention, LN, trunk/dir/gate. Writes out[b*4+n].
// ---------------------------------------------------------------------------
__global__ void regime_head_kernel(const float* __restrict__ h,
    const float* rd1W, const float* rd1b, const float* rd2W, const float* rd2b,
    const float* gnw, const float* gnb, const float* g1W, const float* g1b,
    const float* g2W, const float* g2b,
    const float* WqW, const float* Wqb, const float* WkW, const float* Wkb,
    const float* WvW, const float* Wvb, const float* nw, const float* nb,
    const float* trW, const float* trb, const float* dirW, const float* dirb,
    const float* gateW, const float* gateb,
    float* __restrict__ out)
{
    __shared__ float st[4][96], gm[96], gs[96];
    __shared__ float rh[4][96], rp[4][3];
    __shared__ float gi[4][291];
    __shared__ float g1h[4][64];
    __shared__ float alpha[4];
    __shared__ float Q[4][96], Kk[4][96], V[4][96];
    __shared__ float attn[4][4];
    __shared__ float gout[4][96];
    __shared__ float trunk[4][48];
    const int b = blockIdx.x, tid = threadIdx.x;

    for (int i = tid; i < 4 * 96; i += blockDim.x) {
        const int n = i / 96, d = i % 96;
        float acc = 0.f;
        for (int s = 253; s < 256; ++s)
            acc += h[(size_t)(((b * 256 + s) * 4 + n)) * 96 + d];
        st[n][d] = acc * (1.f / 3.f);
    }
    __syncthreads();
    for (int d = tid; d < 96; d += blockDim.x) {
        const float m = 0.25f * (st[0][d] + st[1][d] + st[2][d] + st[3][d]);
        gm[d] = m;
        float v = 0.f;
        for (int n = 0; n < 4; ++n) { float dd = st[n][d] - m; v += dd * dd; }
        gs[d] = sqrtf(v * (1.f / 3.f));                      // ddof=1
    }
    __syncthreads();
    for (int i = tid; i < 4 * 96; i += blockDim.x) {         // rd1 + silu
        const int n = i / 96, o = i % 96;
        float acc = rd1b[o];
        const float* w = rd1W + o * 192;
        for (int j = 0; j < 96; ++j) acc += w[j] * st[n][j];
        for (int j = 0; j < 96; ++j) acc += w[96 + j] * gm[j];
        rh[n][o] = siluf_(acc);
    }
    __syncthreads();
    for (int i = tid; i < 4 * 3; i += blockDim.x) {          // rd2
        const int n = i / 3, o = i % 3;
        float acc = rd2b[o];
        const float* w = rd2W + o * 96;
        for (int j = 0; j < 96; ++j) acc += w[j] * rh[n][j];
        rp[n][o] = acc;
    }
    __syncthreads();
    if (tid < 4) {                                           // softmax over 3
        const int n = tid;
        const float mx = fmaxf(rp[n][0], fmaxf(rp[n][1], rp[n][2]));
        const float e0 = __expf(rp[n][0] - mx), e1 = __expf(rp[n][1] - mx), e2 = __expf(rp[n][2] - mx);
        const float inv = 1.f / (e0 + e1 + e2);
        rp[n][0] = e0 * inv; rp[n][1] = e1 * inv; rp[n][2] = e2 * inv;
    }
    __syncthreads();
    for (int i = tid; i < 4 * 291; i += blockDim.x) {        // gi concat
        const int n = i / 291, j = i % 291;
        gi[n][j] = (j < 96) ? st[n][j] : (j < 192) ? gm[j - 96]
                 : (j < 288) ? gs[j - 192] : rp[n][j - 288];
    }
    __syncthreads();
    if (tid < 4) {                                           // gate_norm LN(291)
        const int n = tid;
        float m = 0.f; for (int j = 0; j < 291; ++j) m += gi[n][j];
        m /= 291.f;
        float v = 0.f; for (int j = 0; j < 291; ++j) { float d = gi[n][j] - m; v += d * d; }
        const float rs = rsqrtf(v / 291.f + 1e-5f);
        for (int j = 0; j < 291; ++j) gi[n][j] = (gi[n][j] - m) * rs * gnw[j] + gnb[j];
    }
    __syncthreads();
    for (int i = tid; i < 4 * 64; i += blockDim.x) {         // g1 relu
        const int n = i / 64, o = i % 64;
        float acc = g1b[o];
        const float* w = g1W + o * 291;
        for (int j = 0; j < 291; ++j) acc += w[j] * gi[n][j];
        g1h[n][o] = fmaxf(acc, 0.f);
    }
    __syncthreads();
    if (tid < 4) {                                           // g2 sigmoid -> alpha
        const int n = tid;
        float acc = g2b[0];
        for (int j = 0; j < 64; ++j) acc += g2W[j] * g1h[n][j];
        alpha[n] = sigmoidf_(acc);
    }
    for (int i = tid; i < 4 * 96; i += blockDim.x) {         // QKV
        const int n = i / 96, o = i % 96;
        float aq = Wqb[o], ak = Wkb[o], av = Wvb[o];
        for (int j = 0; j < 96; ++j) {
            const float sv = st[n][j];
            aq += WqW[o * 96 + j] * sv; ak += WkW[o * 96 + j] * sv; av += WvW[o * 96 + j] * sv;
        }
        Q[n][o] = aq; Kk[n][o] = ak; V[n][o] = av;
    }
    __syncthreads();
    if (tid < 16) {                                          // QK^T / sqrt(d)
        const int n = tid / 4, m2 = tid % 4;
        float acc = 0.f;
        for (int j = 0; j < 96; ++j) acc += Q[n][j] * Kk[m2][j];
        attn[n][m2] = acc * (1.f / sqrtf(96.f));
    }
    __syncthreads();
    if (tid < 4) {                                           // softmax + alpha mix
        const int n = tid;
        float mx = attn[n][0];
        for (int m2 = 1; m2 < 4; ++m2) mx = fmaxf(mx, attn[n][m2]);
        float s = 0.f;
        for (int m2 = 0; m2 < 4; ++m2) { attn[n][m2] = __expf(attn[n][m2] - mx); s += attn[n][m2]; }
        const float inv = 1.f / s, a = alpha[n];
        for (int m2 = 0; m2 < 4; ++m2)
            attn[n][m2] = (1.f - a) * attn[n][m2] * inv + ((m2 == n) ? a : 0.f);
    }
    __syncthreads();
    for (int i = tid; i < 4 * 96; i += blockDim.x) {         // mixed@V + residual
        const int n = i / 96, d = i % 96;
        float acc = st[n][d];
        for (int m2 = 0; m2 < 4; ++m2) acc += attn[n][m2] * V[m2][d];
        gout[n][d] = acc;
    }
    __syncthreads();
    if (tid < 4) {                                           // norm LN(96)
        const int n = tid;
        float m = 0.f; for (int j = 0; j < 96; ++j) m += gout[n][j];
        m *= (1.f / 96.f);
        float v = 0.f; for (int j = 0; j < 96; ++j) { float d = gout[n][j] - m; v += d * d; }
        const float rs = rsqrtf(v * (1.f / 96.f) + 1e-5f);
        for (int j = 0; j < 96; ++j) gout[n][j] = (gout[n][j] - m) * rs * nw[j] + nb[j];
    }
    __syncthreads();
    for (int i = tid; i < 4 * 48; i += blockDim.x) {         // trunk gelu(exact)
        const int n = i / 48, o = i % 48;
        float acc = trb[o];
        for (int j = 0; j < 96; ++j) acc += trW[o * 96 + j] * gout[n][j];
        trunk[n][o] = 0.5f * acc * (1.f + erff(acc * 0.7071067811865476f));
    }
    __syncthreads();
    if (tid < 4) {                                           // dir*gate
        const int n = tid;
        float ad = dirb[0], ag = gateb[0];
        for (int j = 0; j < 48; ++j) { ad += dirW[j] * trunk[n][j]; ag += gateW[j] * trunk[n][j]; }
        out[b * 4 + n] = tanhf(ad) * sigmoidf_(ag);
    }
}

// ---------------------------------------------------------------------------
// Host orchestration
// d_in leaf order (setup_inputs dict-insertion order):
//  0 x | 1-2 emb1.W/b | 3-4 emb2.W/b | 5-6 input_norm.w/b | 7 pos_emb
//  8..26  delta[0]: Wq Wk Wv vg1.W/b vg2.W/b eta1.W/b eta2.W/b al1.W/b al2.W/b out.W/b norm.w/b
//  27..45 delta[1]: same
//  46..57 cms mlps (l1.W/b l2.W/b) x3 | 58 level_weights | 59..64 level_norms | 65-66 final_norm
//  67..84 regime | 85-86 trunk | 87-88 dir | 89-90 gate
// ---------------------------------------------------------------------------
extern "C" void kernel_launch(void* const* d_in, const int* in_sizes, int n_in,
                              void* d_out, int out_size, void* d_ws, size_t ws_size,
                              hipStream_t stream) {
    (void)in_sizes; (void)n_in; (void)out_size; (void)ws_size;
    auto P = [&](int i) -> const float* { return (const float*)d_in[i]; };

    // ---- workspace bump allocator ----
    size_t off = 0;
    auto balloc = [&](size_t bytes) -> void* {
        void* p = (char*)d_ws + off;
        off += bytes;
        off = (off + 255) & ~(size_t)255;
        return p;
    };
    float*  hF   = (float*)balloc((size_t)ROWS * DMODEL * 4);
    __bf16* hB   = (__bf16*)balloc((size_t)ROWS * DMODEL * 2);
    float*  bufQ = (float*)balloc((size_t)ROWS * DMODEL * 4);
    float*  bufK = (float*)balloc((size_t)ROWS * DMODEL * 4);
    float*  bufV = (float*)balloc((size_t)ROWS * DMODEL * 4);
    float*  bufO = (float*)balloc((size_t)ROWS * DMODEL * 4);
    __bf16* bfA  = (__bf16*)balloc((size_t)ROWS * DMODEL * 2);
    __bf16* bfB  = (__bf16*)balloc((size_t)ROWS * DMODEL * 2);
    float*  etaB = (float*)balloc((size_t)ROWS * 4);
    float*  alB  = (float*)balloc((size_t)ROWS * 4);
    __bf16* t288 = (__bf16*)balloc((size_t)ROWS * 288 * 2);

    __bf16 *WqB[2], *WkB[2], *WvB[2], *Vg1B[2], *Vg2B[2], *OutB[2];
    __bf16 *L1B[3], *L2B[3];
    for (int l = 0; l < 2; ++l) {
        WqB[l]  = (__bf16*)balloc(9216 * 2); WkB[l]  = (__bf16*)balloc(9216 * 2);
        WvB[l]  = (__bf16*)balloc(9216 * 2); Vg1B[l] = (__bf16*)balloc(9216 * 2);
        Vg2B[l] = (__bf16*)balloc(9216 * 2); OutB[l] = (__bf16*)balloc(9216 * 2);
    }
    for (int i = 0; i < 3; ++i) {
        L1B[i] = (__bf16*)balloc(27648 * 2);
        L2B[i] = (__bf16*)balloc(27648 * 2);
    }

    auto conv = [&](const float* src, __bf16* dst, int n) {
        f32_to_bf16_kernel<<<dim3((n + 255) / 256), dim3(256), 0, stream>>>(src, dst, n);
    };
    for (int l = 0; l < 2; ++l) {
        const int pb = 8 + l * 19;
        conv(P(pb + 0),  WqB[l],  9216);
        conv(P(pb + 1),  WkB[l],  9216);
        conv(P(pb + 2),  WvB[l],  9216);
        conv(P(pb + 3),  Vg1B[l], 9216);
        conv(P(pb + 5),  Vg2B[l], 9216);
        conv(P(pb + 15), OutB[l], 9216);
    }
    for (int i = 0; i < 3; ++i) {
        conv(P(46 + i * 4 + 0), L1B[i], 27648);
        conv(P(46 + i * 4 + 2), L2B[i], 27648);
    }

    auto gemm = [&](const __bf16* A, const __bf16* W, const float* bias,
                    float* Cf, __bf16* Cb, int N, int K, int act) {
        dim3 g(ROWS / 32, (N + 63) / 64);
        gemm_wmma_bf16<<<g, dim3(128), 0, stream>>>(A, W, bias, Cf, Cb, ROWS, N, K, act);
    };
    const dim3 rowsGrid(ROWS / 8);     // wave-per-row kernels, 256 thr = 8 waves
    const dim3 rowsBlk(256);

    // ---- embed + pos + input LN ----
    embed_kernel<<<dim3(ROWS), dim3(96), 0, stream>>>(
        P(0), P(1), P(2), P(3), P(4), P(5), P(6), P(7), hF, hB);

    // ---- two delta layers ----
    for (int l = 0; l < 2; ++l) {
        const int pb = 8 + l * 19;
        gemm(hB, WqB[l], nullptr, bufQ, nullptr, 96, 96, 0);         // q
        gemm(hB, WkB[l], nullptr, bufK, nullptr, 96, 96, 0);         // k
        gemm(hB, WvB[l], nullptr, nullptr, bfA, 96, 96, 0);          // v (bf16)
        knorm_kernel<<<rowsGrid, rowsBlk, 0, stream>>>(bufK, ROWS);  // kn
        eta_al_kernel<<<dim3(ROWS / 256), dim3(256), 0, stream>>>(
            hF, P(pb + 7), P(pb + 8), P(pb + 9), P(pb + 10),
            P(pb + 11), P(pb + 12), P(pb + 13), P(pb + 14), etaB, alB, ROWS);
        gemm(bfA, Vg1B[l], P(pb + 4), nullptr, bfB, 96, 96, 1);      // silu(vg1)
        gemm(bfB, Vg2B[l], P(pb + 6), bufV, nullptr, 96, 96, 0);     // vh

        float* Mout = (float*)d_out + 128 + (size_t)l * 128 * 96 * 96;
        delta_scan_kernel<<<dim3(128), dim3(96), 0, stream>>>(
            bufQ, bufK, bufV, etaB, alB, bufO, bfA, Mout);

        gemm(bfA, OutB[l], P(pb + 16), bufK, nullptr, 96, 96, 0);    // out proj
        add_ln_kernel<<<rowsGrid, rowsBlk, 0, stream>>>(
            bufK, hF, P(pb + 17), P(pb + 18), hF, hB, ROWS);         // h = LN(out + h)
    }

    // ---- CMS ----
    float* outs[3] = { bufQ, bufK, bufV };
    for (int lv = 0; lv < 3; ++lv) {
        gemm(hB, L1B[lv], P(46 + lv * 4 + 1), nullptr, t288, 288, 96, 1);
        gemm(t288, L2B[lv], P(46 + lv * 4 + 3), bufO, nullptr, 96, 288, 0);
        add_ln_kernel<<<rowsGrid, rowsBlk, 0, stream>>>(
            bufO, hF, P(59 + lv * 2), P(60 + lv * 2), outs[lv], nullptr, ROWS);
    }
    cms_combine_kernel<<<dim3((ROWS * 96 + 255) / 256), dim3(256), 0, stream>>>(
        outs[0], outs[1], outs[2], P(58), bufO, ROWS * 96);
    add_ln_kernel<<<rowsGrid, rowsBlk, 0, stream>>>(
        bufO, nullptr, P(65), P(66), hF, nullptr, ROWS);

    // ---- regime + head -> d_out[0..127] ----
    regime_head_kernel<<<dim3(32), dim3(128), 0, stream>>>(
        hF,
        P(67), P(68), P(69), P(70), P(71), P(72), P(73), P(74), P(75), P(76),
        P(77), P(78), P(79), P(80), P(81), P(82), P(83), P(84),
        P(85), P(86), P(87), P(88), P(89), P(90),
        (float*)d_out);
}
